// IrrepsBlock_64742337020477
// MI455X (gfx1250) — compile-verified, matrix operand order checked by reading, not run on hardware
//
#include <hip/hip_runtime.h>

typedef __attribute__((ext_vector_type(16))) __bf16 v16bf;
typedef __attribute__((ext_vector_type(8)))  float  v8f;
typedef __attribute__((ext_vector_type(4)))  unsigned int v4u;
typedef __attribute__((ext_vector_type(8)))  int v8i;
typedef __attribute__((ext_vector_type(4)))  int v4i;

// ---------------- constants -----------------------------------------------
#define NALL   1024
#define EDGES  32768
#define YW     3136           // y width
#define NODE_ELEMS (NALL * YW)
#define YOUT_OFF   (NALL * YW)   // y_out starts here in d_out

// fragment = 32 lanes x 16 bf16 = 512 bf16 (1 KB)
#define FRAG_BF16 512
// packed-weight fragment offsets (in fragments)
constexpr int SIB[3][3]   = {{0,16,24},{32,40,44},{48,52,54}};
constexpr int LIN0F = 56;   // KF=7 , NF=8
constexpr int LIN1F = 112;  // KF=12, NF=2
constexpr int LIN2F = 136;  // KF=11, NF=2
#define TOTAL_FRAGS 158

// ---------------- W3J sparse tables ---------------------------------------
struct Nz { int i, j, k; float v; };
constexpr float KA  = 0.31622777f;  // 1/sqrt(10)
constexpr float KB_ = 0.36514837f;  // 2/sqrt(30)
constexpr float KC  = 0.18257419f;  // 1/sqrt(30)
constexpr float K3  = 0.57735027f;  // 1/sqrt(3)
constexpr float K5  = 0.44721360f;  // 1/sqrt(5)
constexpr float K6  = 0.40824829f;  // 1/sqrt(6)
constexpr float KP  = 0.20701967f;  // sqrt(3/70)
constexpr float KQ  = 0.23904572f;  // 2/sqrt(70)
constexpr float KR  = 0.11952286f;  // 1/sqrt(70)

constexpr Nz W3NZ[137] = {
  // C000 o=0 (1)
  {0,0,0, 1.0f},
  // C011 o=1 (3)
  {0,0,0,K3},{0,1,1,K3},{0,2,2,K3},
  // C022 o=4 (5)
  {0,0,0,K5},{0,1,1,K5},{0,2,2,K5},{0,3,3,K5},{0,4,4,K5},
  // C110 o=9 (3)
  {0,0,0,K3},{1,1,0,K3},{2,2,0,K3},
  // C101 o=12 (3)
  {0,0,0,K3},{1,0,1,K3},{2,0,2,K3},
  // C111 o=15 (6)  epsilon/sqrt6
  {0,1,2,K6},{1,2,0,K6},{2,0,1,K6},{0,2,1,-K6},{2,1,0,-K6},{1,0,2,-K6},
  // C121 o=21 (11)
  {2,0,0,KA},{0,0,2,KA},{0,1,1,KA},{1,1,0,KA},{1,2,1,KB_},
  {0,2,0,-KC},{2,2,2,-KC},{2,3,1,KA},{1,3,2,KA},{2,4,2,KA},{0,4,0,-KA},
  // C112 o=32 (11)
  {2,0,0,KA},{0,2,0,KA},{0,1,1,KA},{1,0,1,KA},{1,1,2,KB_},
  {0,0,2,-KC},{2,2,2,-KC},{2,1,3,KA},{1,2,3,KA},{2,2,4,KA},{0,0,4,-KA},
  // C122 o=43 (16)  generators L_a / sqrt(30)
  {0,1,0,KC},{0,0,1,-KC},{0,3,2,-KA},{0,2,3,KA},{0,4,3,-KC},{0,3,4,KC},
  {1,4,0,KB_},{1,0,4,-KB_},{1,3,1,KC},{1,1,3,-KC},
  {2,3,0,-KC},{2,0,3,KC},{2,4,1,-KC},{2,1,4,KC},{2,2,1,-KA},{2,1,2,KA},
  // C220 o=59 (5)
  {0,0,0,K5},{1,1,0,K5},{2,2,0,K5},{3,3,0,K5},{4,4,0,K5},
  // C211 o=64 (11)
  {0,2,0,KA},{0,0,2,KA},{1,0,1,KA},{1,1,0,KA},{2,1,1,KB_},
  {2,0,0,-KC},{2,2,2,-KC},{3,2,1,KA},{3,1,2,KA},{4,2,2,KA},{4,0,0,-KA},
  // C221 o=75 (16)
  {1,0,0,KC},{0,1,0,-KC},{3,2,0,-KA},{2,3,0,KA},{4,3,0,-KC},{3,4,0,KC},
  {4,0,1,KB_},{0,4,1,-KB_},{3,1,1,KC},{1,3,1,-KC},
  {3,0,2,-KC},{0,3,2,KC},{4,1,2,-KC},{1,4,2,KC},{2,1,2,-KA},{1,2,2,KA},
  // C202 o=91 (5)
  {0,0,0,K5},{1,0,1,K5},{2,0,2,K5},{3,0,3,K5},{4,0,4,K5},
  // C212 o=96 (16)
  {1,0,0,-KC},{0,0,1,KC},{3,0,2,KA},{2,0,3,-KA},{4,0,3,KC},{3,0,4,-KC},
  {4,1,0,-KB_},{0,1,4,KB_},{3,1,1,-KC},{1,1,3,KC},
  {3,2,0,KC},{0,2,3,-KC},{4,2,1,KC},{1,2,4,-KC},{2,2,1,KA},{1,2,2,-KA},
  // C222 o=112 (25)
  {0,1,3,KP},{0,3,1,KP},{1,0,3,KP},{1,3,0,KP},{3,0,1,KP},{3,1,0,KP},
  {0,0,2,-KQ},{0,2,0,-KQ},{2,0,0,-KQ},
  {1,1,2,KR},{1,2,1,KR},{2,1,1,KR},
  {3,3,2,KR},{3,2,3,KR},{2,3,3,KR},
  {1,1,4,-KP},{1,4,1,-KP},{4,1,1,-KP},
  {3,3,4,KP},{3,4,3,KP},{4,3,3,KP},
  {2,2,2,KQ},
  {2,4,4,-KQ},{4,2,4,-KQ},{4,4,2,-KQ},
};

struct TpI { int l1, l2, l3, nzo, nzc, tpRow, linKB, yBase, U; float scale; };
constexpr TpI INSTRS[15] = {
  {0,0,0,  0, 1, 0, 0,    0, 128, 0.125f},
  {0,1,1,  1, 3, 1, 0,  128, 128, 0.30618622f},
  {0,2,2,  4, 5, 2, 0,  512, 128, 0.39528471f},
  {1,1,0,  9, 3, 0, 4, 1152,  64, 0.17677670f},
  {1,0,1, 12, 3, 1, 4, 1216,  64, 0.21650635f},
  {1,1,1, 15, 6, 2, 6, 1408,  64, 0.30618622f},
  {1,2,1, 21,11, 3, 8, 1600,  64, 0.30618622f},
  {1,1,2, 32,11, 4, 4, 1792,  64, 0.39528471f},
  {1,2,2, 43,16, 5, 6, 2112,  64, 0.39528471f},
  {2,2,0, 59, 5, 0, 6, 2432,  32, 0.17677670f},
  {2,1,1, 64,11, 1,10, 2464,  32, 0.30618622f},
  {2,2,1, 75,16, 2,11, 2560,  32, 0.30618622f},
  {2,0,2, 91, 5, 3, 8, 2656,  32, 0.27950850f},
  {2,1,2, 96,16, 4, 9, 2816,  32, 0.39528471f},
  {2,2,2,112,25, 5,10, 2976,  32, 0.39528471f},
};

// ---------------- WMMA helper ---------------------------------------------
__device__ inline v8f wmma_bf16(v16bf a, v16bf b, v8f c) {
  return __builtin_amdgcn_wmma_f32_16x16x32_bf16(false, a, false, b, (short)0, c,
                                                 false, false);
}

// ---------------- prep kernel: zero node out + pack weights to bf16 frags --
__global__ void prep_pack_zero(
    const float* w0, const float* w1, const float* w2, const float* w3,
    const float* w4, const float* w5, const float* w6, const float* w7,
    const float* w8, const float* w9, const float* w10, const float* w11,
    unsigned short* wsb, float* out)
{
  const int tid = blockIdx.x * blockDim.x + threadIdx.x;
  const int stride = gridDim.x * blockDim.x;
  for (int i = tid; i < NODE_ELEMS; i += stride) out[i] = 0.0f;

  const float* Wp[12] = {w0,w1,w2,w3,w4,w5,w6,w7,w8,w9,w10,w11};
  const int KFa[12] = {2,1,1, 2,1,1, 2,1,1, 7,12,11};
  const int NFa[12] = {8,8,8, 4,4,4, 2,2,2, 8,2,2};
  const int OFF[13] = {0,16,24,32,40,44,48,52,54,56,112,136,158};
  __bf16* wb = (__bf16*)wsb;
  for (int idx = tid; idx < TOTAL_FRAGS * FRAG_BF16; idx += stride) {
    int frag   = idx >> 9;
    int within = idx & 511;
    int lane   = within >> 4;
    int q      = within & 15;
    int m = 0;
#pragma unroll
    for (int j = 0; j < 12; ++j) if (frag >= OFF[j + 1]) m = j + 1;
    int lf = frag - OFF[m];
    int kfr = KFa[m];
    int nf = lf / kfr, kf = lf % kfr;
    int n = nf * 16 + (lane & 15);
    int k = kf * 32 + (lane >> 4) * 16 + q;
    wb[idx] = (__bf16)Wp[m][k * (NFa[m] * 16) + n];
  }
}

// ---------------- per-instruction fused TP + lin accumulate ----------------
template <int IT>
__device__ __forceinline__ void do_instr(
    int lane, int e0, int dstv, float invDen,
    const float* __restrict__ tpw0, const float* __restrict__ tpw1,
    const float* __restrict__ tpw2,
    const v16bf* __restrict__ WSF, float* __restrict__ out,
    float (&xsLds)[16][480], __bf16 (&x2Lds)[16][320],
    __bf16 (&stage)[5][16][32],
    v8f (&accH0)[8], v8f (&accH1)[6], v8f (&accH2)[10])
{
  constexpr TpI ins = INSTRS[IT];
  constexpr int l1 = ins.l1;
  constexpr int d1 = 2 * l1 + 1;
  constexpr int d2 = 2 * ins.l2 + 1;
  constexpr int d3 = 2 * ins.l3 + 1;
  constexpr int nfCount = ins.U / 16;
  constexpr int kfr  = (ins.l2 == 0) ? 2 : 1;
  constexpr int fb   = SIB[l1][ins.l2];
  constexpr int strd = (ins.l2 == 0) ? 1 : (ins.l2 == 1 ? 3 : 5);
  constexpr int cb0  = (ins.l2 == 0) ? 0 : (ins.l2 == 1 ? 64 : 160);

  const float* tpw = (l1 == 0) ? tpw0 : (l1 == 1) ? tpw1 : tpw2;
  const int laneLo = lane & 15;
  const int laneHi = lane >> 4;

  // ---- hoisted A fragments (independent of output-channel fragment nf) ----
  v16bf afr[d2][kfr];
#pragma unroll
  for (int m = 0; m < d2; ++m)
#pragma unroll
    for (int kf = 0; kf < kfr; ++kf) {
      const int k0 = kf * 32 + laneHi * 16;
#pragma unroll
      for (int q = 0; q < 16; ++q)
        afr[m][kf][q] = x2Lds[laneLo][cb0 + m + (k0 + q) * strd];
    }

  for (int nf = 0; nf < nfCount; ++nf) {
    // ---- si GEMM: kb components (D fragment layout: lane<->(edge,channel))
    v8f kb[5];
#pragma unroll
    for (int m = 0; m < d2; ++m) {
      v8f acc = {};
#pragma unroll
      for (int kf = 0; kf < kfr; ++kf) {
        v16bf b = WSF[(fb + nf * kfr + kf) * 32 + lane];
        acc = wmma_bf16(afr[m][kf], b, acc);
      }
      kb[m] = acc;
    }

    // ---- gather x components at this lane's (edge,channel) positions
    const int u16 = nf * 16 + laneLo;
    float X[d1][8];
#pragma unroll
    for (int r = 0; r < 8; ++r) {
      int erow = r + 8 * laneHi;
      int cbx = (l1 == 0) ? u16 : (l1 == 1 ? 128 + u16 * 3 : 320 + u16 * 5);
#pragma unroll
      for (int i = 0; i < d1; ++i) X[i][r] = xsLds[erow][cbx + i];
    }

    const float tw = tpw[ins.tpRow * ins.U + u16] * ins.scale;

    // ---- sparse W3J contraction
    float Y[d3][8] = {};
#pragma unroll
    for (int t = 0; t < ins.nzc; ++t) {
      const int ci = W3NZ[ins.nzo + t].i;
      const int cj = W3NZ[ins.nzo + t].j;
      const int ck = W3NZ[ins.nzo + t].k;
      const float cv = W3NZ[ins.nzo + t].v;
#pragma unroll
      for (int r = 0; r < 8; ++r) Y[ck][r] += cv * X[ci][r] * kb[cj][r];
    }

    // ---- scatter-add into node sums; stage bf16 for second linear
#pragma unroll
    for (int k = 0; k < d3; ++k) {
#pragma unroll
      for (int r = 0; r < 8; ++r) {
        float yv = Y[k][r] * tw;
        int erow = r + 8 * laneHi;
        int dnode = __shfl(dstv, erow);
        unsafeAtomicAdd(out + (size_t)dnode * YW + ins.yBase + u16 * d3 + k,
                        yv * invDen);
        stage[k][erow][(nf & 1) * 16 + laneLo] = (__bf16)yv;
      }
    }

    // ---- every 32 channels: WMMA-accumulate into second-linear outputs
    if (nf & 1) {
      asm volatile("s_wait_dscnt 0" ::: "memory");
      const int kblk = ins.linKB + (nf >> 1);
      if (ins.l3 == 0) {
        v16bf a = *(const v16bf*)&stage[0][laneLo][laneHi * 16];
#pragma unroll
        for (int f = 0; f < 8; ++f)
          accH0[f] = wmma_bf16(a, WSF[(LIN0F + f * 7 + kblk) * 32 + lane], accH0[f]);
      } else if (ins.l3 == 1) {
#pragma unroll
        for (int m = 0; m < 3; ++m) {
          v16bf a = *(const v16bf*)&stage[m][laneLo][laneHi * 16];
#pragma unroll
          for (int f = 0; f < 2; ++f)
            accH1[m * 2 + f] =
                wmma_bf16(a, WSF[(LIN1F + f * 12 + kblk) * 32 + lane], accH1[m * 2 + f]);
        }
      } else {
#pragma unroll
        for (int m = 0; m < 5; ++m) {
          v16bf a = *(const v16bf*)&stage[m][laneLo][laneHi * 16];
#pragma unroll
          for (int f = 0; f < 2; ++f)
            accH2[m * 2 + f] =
                wmma_bf16(a, WSF[(LIN2F + f * 11 + kblk) * 32 + lane], accH2[m * 2 + f]);
        }
      }
    }
  }
}

// ---------------- main fused kernel: 1 wave/block, 16 edges/wave -----------
__global__ __launch_bounds__(32) void irreps_main(
    const float* __restrict__ x1, const float* __restrict__ x2,
    const int* __restrict__ ei,
    const float* __restrict__ tp0w, const float* __restrict__ tp1w,
    const float* __restrict__ tp2w,
    const float* __restrict__ den,
    const unsigned short* __restrict__ wsb, float* __restrict__ out)
{
  const int lane = threadIdx.x & 31;
  const int e0   = blockIdx.x * 16;
  const int laneLo = lane & 15;
  const int laneHi = lane >> 4;

  __shared__ __align__(64) float  xsLds[16][480];   // f32 rows via TDM gather
  __shared__ __align__(64) __bf16 x2Lds[16][320];   // edge features (bf16)
  __shared__ __align__(64) __bf16 stage[5][16][32]; // y transpose staging

  // -------- TDM gather: 16 src node rows (480 f32 each) -> LDS -------------
  int srcIdx[16];
#pragma unroll
  for (int r = 0; r < 16; ++r) srcIdx[r] = ei[2 * (e0 + r) + 1];  // uniform

  {
    unsigned long long ga = (unsigned long long)(const void*)x1;
    unsigned ldsAddr = (unsigned)(unsigned long long)(void*)&xsLds[0][0];
    // D# group0: count=1, gather_mode=1, 16-bit indices, type=2
    v4u g0 = { 1u | 0x80000000u, ldsAddr,
               (unsigned)(ga & 0xffffffffull),
               (unsigned)((ga >> 32) & 0x1ffffffull) | (2u << 30) };
    // D# group1: data_size=4B, tensor_dim0=480, tensor_dim1=1024,
    //            tile_dim0=480, tile_dim1=16 indices, dim0_stride=480
    v8i g1 = { (int)(2u << 16),
               (int)(480u << 16),
               (int)(1024u << 16),
               (int)(480u << 16),
               16, 480, 0, 0 };
    // groups 2/3: sixteen 16-bit row indices
    v4i g2 = { srcIdx[0]  | (srcIdx[1]  << 16), srcIdx[2]  | (srcIdx[3]  << 16),
               srcIdx[4]  | (srcIdx[5]  << 16), srcIdx[6]  | (srcIdx[7]  << 16) };
    v4i g3 = { srcIdx[8]  | (srcIdx[9]  << 16), srcIdx[10] | (srcIdx[11] << 16),
               srcIdx[12] | (srcIdx[13] << 16), srcIdx[14] | (srcIdx[15] << 16) };
    v8i g4 = { 0, 0, 0, 0, 0, 0, 0, 0 };
    __builtin_amdgcn_tensor_load_to_lds(g0, g1, g2, g3, g4, 0);
  }

  // -------- stage x2 tile to LDS (bf16), overlapped with TDM ---------------
  for (int idx = lane; idx < 16 * 320; idx += 32) {
    int rr = idx / 320, c = idx - rr * 320;
    x2Lds[rr][c] = (__bf16)x2[(size_t)(e0 + rr) * 320 + c];
  }

  // dst indices for the atomic scatter (per-lane, broadcast via shfl)
  int dstv = 0;
  if (lane < 16) dstv = ei[2 * (e0 + lane)];

  const float invDen = 1.0f / den[0];
  const v16bf* WSF = (const v16bf*)wsb;

  // wait for TDM gather before TP reads xsLds
  __builtin_amdgcn_s_wait_tensorcnt(0);
  asm volatile("s_wait_dscnt 0" ::: "memory");

  v8f accH0[8]  = {};
  v8f accH1[6]  = {};
  v8f accH2[10] = {};

#define CALL(n) do_instr<n>(lane, e0, dstv, invDen, tp0w, tp1w, tp2w, WSF, \
                            out, xsLds, x2Lds, stage, accH0, accH1, accH2)
  CALL(0);  CALL(1);  CALL(2);
  CALL(3);  CALL(4);  CALL(5);  CALL(6);  CALL(7);  CALL(8);
  CALL(9);  CALL(10); CALL(11); CALL(12); CALL(13); CALL(14);
#undef CALL

  // ---- epilogue: scale, gated SiLU, store y_out --------------------------
  const float inv224 = 0.066815310f;
  const float inv384 = 0.051031036f;
  const float inv352 = 0.053300179f;
  const float SILU_C = 1.6790f;
  float* yo = out + YOUT_OFF;

#pragma unroll
  for (int f = 0; f < 4; ++f)
#pragma unroll
    for (int r = 0; r < 8; ++r) {
      float h = accH0[f][r] * inv224;
      float sv = SILU_C * h / (1.0f + __expf(-h));
      int e = e0 + r + 8 * laneHi;
      yo[(size_t)e * 320 + f * 16 + laneLo] = sv;
    }

  float G1v[2][8], G2v[2][8];
#pragma unroll
  for (int f = 0; f < 2; ++f)
#pragma unroll
    for (int r = 0; r < 8; ++r) {
      float ha = accH0[4 + f][r] * inv224;
      G1v[f][r] = SILU_C * ha / (1.0f + __expf(-ha));
      float hb = accH0[6 + f][r] * inv224;
      G2v[f][r] = SILU_C * hb / (1.0f + __expf(-hb));
    }

#pragma unroll
  for (int m = 0; m < 3; ++m)
#pragma unroll
    for (int f = 0; f < 2; ++f)
#pragma unroll
      for (int r = 0; r < 8; ++r) {
        int e = e0 + r + 8 * laneHi;
        int u = f * 16 + laneLo;
        yo[(size_t)e * 320 + 64 + u * 3 + m] = accH1[m * 2 + f][r] * inv384 * G1v[f][r];
      }
#pragma unroll
  for (int m = 0; m < 5; ++m)
#pragma unroll
    for (int f = 0; f < 2; ++f)
#pragma unroll
      for (int r = 0; r < 8; ++r) {
        int e = e0 + r + 8 * laneHi;
        int u = f * 16 + laneLo;
        yo[(size_t)e * 320 + 160 + u * 5 + m] = accH2[m * 2 + f][r] * inv352 * G2v[f][r];
      }
}

// ---------------- launch ----------------------------------------------------
extern "C" void kernel_launch(void* const* d_in, const int* in_sizes, int n_in,
                              void* d_out, int out_size, void* d_ws, size_t ws_size,
                              hipStream_t stream) {
  (void)in_sizes; (void)n_in; (void)out_size; (void)ws_size;
  const float* x1   = (const float*)d_in[0];
  const float* x2   = (const float*)d_in[1];
  const int*   ei   = (const int*)d_in[2];
  const float* si1w0 = (const float*)d_in[3];
  const float* si1w1 = (const float*)d_in[4];
  const float* si1w2 = (const float*)d_in[5];
  const float* si2w0 = (const float*)d_in[6];
  const float* si2w1 = (const float*)d_in[7];
  const float* si2w2 = (const float*)d_in[8];
  const float* si3w0 = (const float*)d_in[9];
  const float* si3w1 = (const float*)d_in[10];
  const float* si3w2 = (const float*)d_in[11];
  const float* tp0w  = (const float*)d_in[12];
  const float* tp1w  = (const float*)d_in[13];
  const float* tp2w  = (const float*)d_in[14];
  const float* linw0 = (const float*)d_in[15];
  const float* linw1 = (const float*)d_in[16];
  const float* linw2 = (const float*)d_in[17];
  const float* den   = (const float*)d_in[18];
  float* out = (float*)d_out;
  unsigned short* wsb = (unsigned short*)d_ws;

  prep_pack_zero<<<2048, 256, 0, stream>>>(si1w0, si1w1, si1w2,
                                           si2w0, si2w1, si2w2,
                                           si3w0, si3w1, si3w2,
                                           linw0, linw1, linw2, wsb, out);
  irreps_main<<<EDGES / 16, 32, 0, stream>>>(x1, x2, ei, tp0w, tp1w, tp2w,
                                             den, wsb, out);
}
